// GAT_24721831756422
// MI455X (gfx1250) — matrix operand order repaired and checked by reading, not compiled, not run on hardware
//
#include <hip/hip_runtime.h>
#include <hip/hip_bf16.h>

#define NNODES 50000
#define NEDGES 1600000
#define ETOT   (NNODES + NEDGES)
#define NFEAT  128
#define TPB    256

typedef __attribute__((ext_vector_type(16))) _Float16 v16h;
typedef __attribute__((ext_vector_type(8)))  _Float16 v8h;
typedef __attribute__((ext_vector_type(8)))  float    v8f;

__device__ __forceinline__ float lrelu(float t) { return t > 0.f ? t : 0.2f * t; }

// Order-independent float atomic max via sign-split integer atomics
// (positive values: signed int max; negative: unsigned min). Deterministic.
__device__ __forceinline__ void atomicMaxF(float* addr, float v) {
  if (v >= 0.f) atomicMax((int*)addr, __float_as_int(v));
  else          atomicMin((unsigned int*)addr, __float_as_uint(v));
}

// ---------------------------------------------------------------- edge setup
__global__ void k_edges(const int* __restrict__ ei, int* __restrict__ src,
                        int* __restrict__ dst) {
  int i = blockIdx.x * blockDim.x + threadIdx.x;
  if (i >= ETOT) return;
  if (i < NEDGES) { src[i] = ei[i]; dst[i] = ei[NEDGES + i]; }
  else            { src[i] = i - NEDGES; dst[i] = i - NEDGES; }   // self-loops
}

__global__ void k_f32tof16(const float* __restrict__ in, _Float16* __restrict__ out, int n) {
  int i = blockIdx.x * blockDim.x + threadIdx.x;
  if (i < n) out[i] = (_Float16)in[i];
}

// Pack W1 [128,48] f32 into WMMA B-matrix (32x16 f16) tile layout:
// flat index = ((kt*3 + nt)*32 + lane)*16 + hh
// element (K = kt*32 + 8*(lane>>4) + (hh&7) + 16*(hh>>3), N = nt*16 + (lane&15))
__global__ void k_packW1(const float* __restrict__ W, _Float16* __restrict__ wp) {
  int i = blockIdx.x * blockDim.x + threadIdx.x;
  if (i >= 4 * 3 * 32 * 16) return;
  int hh   = i & 15;
  int lane = (i >> 4) & 31;
  int t    = i >> 9;
  int nt   = t % 3;
  int kt   = t / 3;
  int k = kt * 32 + 8 * (lane >> 4) + (hh & 7) + 16 * (hh >> 3);
  int n = nt * 16 + (lane & 15);
  wp[i] = (_Float16)W[k * 48 + n];
}

// ------------------------------------------- layer-1 GEMM: [50000,128]x[128,48]
// One wave per 16-row M-tile; 3 N-tiles of 16; K looped in 4 chunks of 32.
__global__ __launch_bounds__(TPB)
void k_gemm_wmma_l1(const _Float16* __restrict__ xh, const _Float16* __restrict__ wp,
                    float* __restrict__ out) {
  int wave = (int)((blockIdx.x * (unsigned)blockDim.x + threadIdx.x) >> 5);
  if (wave >= NNODES / 16) return;            // wave-uniform exit: EXEC stays all-1s
  int lane = threadIdx.x & 31;
  int g = lane >> 4;                          // lane half-group
  int r = lane & 15;
  int m0 = wave * 16;
  v8f c0 = {}, c1 = {}, c2 = {};
  // A layout (16-bit 16x32): halves 0-7 -> cols 8g..8g+7 ; halves 8-15 -> cols 8g+16..8g+23
  const _Float16* arow = xh + (size_t)(m0 + r) * NFEAT + 8 * g;
  for (int kt = 0; kt < 4; ++kt) {
    v8h alo = *(const v8h*)(arow + kt * 32);
    v8h ahi = *(const v8h*)(arow + kt * 32 + 16);
    v16h a;
#pragma unroll
    for (int q = 0; q < 8; ++q) { a[q] = alo[q]; a[q + 8] = ahi[q]; }
    const _Float16* wpt = wp + (size_t)kt * 3 * 512 + lane * 16;
    v16h b0 = *(const v16h*)(wpt);
    v16h b1 = *(const v16h*)(wpt + 512);
    v16h b2 = *(const v16h*)(wpt + 1024);
    c0 = __builtin_amdgcn_wmma_f32_16x16x32_f16(false, a, false, b0, (short)0, c0, false, false);
    c1 = __builtin_amdgcn_wmma_f32_16x16x32_f16(false, a, false, b1, (short)0, c1, false, false);
    c2 = __builtin_amdgcn_wmma_f32_16x16x32_f16(false, a, false, b2, (short)0, c2, false, false);
  }
  // C/D layout: VGPR rr, lanes0-15: (M=rr, N=lane); lanes16-31: (M=8+rr, N=lane-16)
#pragma unroll
  for (int rr = 0; rr < 8; ++rr) {
    float* pdst = out + (size_t)(m0 + rr + 8 * g) * 48 + r;
    pdst[0]  = c0[rr];
    pdst[16] = c1[rr];
    pdst[32] = c2[rr];
  }
}

// -------------------------------------------------- small GEMMs (K = 8 or 16)
__global__ void k_gemm_small(const float* __restrict__ hin, const float* __restrict__ W,
                             float* __restrict__ out, int K, int NC) {
  int i = blockIdx.x * blockDim.x + threadIdx.x;
  if (i >= NNODES * NC) return;
  int n = i / NC, j = i - n * NC;
  const float* row = hin + (size_t)n * K;
  float acc = 0.f;
  for (int k = 0; k < K; ++k) acc += row[k] * W[k * NC + j];
  out[i] = acc;
}

// ------------------------------------------------------------ attention pass
__global__ void k_scores(const float* __restrict__ feat, const float* __restrict__ as_,
                         const float* __restrict__ ad_, float* __restrict__ es,
                         float* __restrict__ ed, float* __restrict__ m,
                         float* __restrict__ s, int H, int C) {
  int i = blockIdx.x * blockDim.x + threadIdx.x;
  if (i >= NNODES * H) return;
  int n = i / H, h = i - n * H;
  const float* f = feat + (size_t)n * H * C + h * C;
  float a = 0.f, b = 0.f;
  for (int c = 0; c < C; ++c) { a += f[c] * as_[h * C + c]; b += f[c] * ad_[h * C + c]; }
  es[i] = a; ed[i] = b; m[i] = -3.0e38f; s[i] = 0.f;
}

__global__ void k_zero(float* __restrict__ p, int n) {
  int i = blockIdx.x * blockDim.x + threadIdx.x;
  if (i < n) p[i] = 0.f;
}

__global__ void k_edge_max(const int* __restrict__ src, const int* __restrict__ dst,
                           const float* __restrict__ es, const float* __restrict__ ed,
                           float* __restrict__ m, int H) {
  int e = blockIdx.x * blockDim.x + threadIdx.x;
  if (e >= ETOT) return;
  int sN = src[e] * H, dN = dst[e] * H;
  for (int h = 0; h < H; ++h) atomicMaxF(&m[dN + h], lrelu(es[sN + h] + ed[dN + h]));
}

__global__ void k_edge_sum(const int* __restrict__ src, const int* __restrict__ dst,
                           const float* __restrict__ es, const float* __restrict__ ed,
                           const float* __restrict__ m, float* __restrict__ s, int H) {
  int e = blockIdx.x * blockDim.x + threadIdx.x;
  if (e >= ETOT) return;
  int sN = src[e] * H, dN = dst[e] * H;
  for (int h = 0; h < H; ++h) {
    float t = lrelu(es[sN + h] + ed[dN + h]);
    atomicAdd(&s[dN + h], expf(t - m[dN + h]));
  }
}

__global__ void k_edge_scatter(const int* __restrict__ src, const int* __restrict__ dst,
                               const float* __restrict__ es, const float* __restrict__ ed,
                               const float* __restrict__ m, const float* __restrict__ ssum,
                               const float* __restrict__ feat, float* __restrict__ accum,
                               int H, int C) {
  long long i = (long long)blockIdx.x * blockDim.x + threadIdx.x;
  if (i >= (long long)ETOT * H) return;
  int e = (int)(i / H), h = (int)(i - (long long)e * H);
  int sN = src[e], dN = dst[e];
  float t = lrelu(es[sN * H + h] + ed[dN * H + h]);
  float alpha = expf(t - m[dN * H + h]) / (ssum[dN * H + h] + 1e-16f);
  const float* fs = feat + (size_t)sN * H * C + h * C;
  float* ac = accum + (size_t)dN * H * C + h * C;
  for (int c = 0; c < C; ++c) atomicAdd(&ac[c], fs[c] * alpha);
}

__global__ void k_head_mean(const float* __restrict__ accum, const float* __restrict__ bias,
                            float* __restrict__ out, int H, int C, int doRelu) {
  int i = blockIdx.x * blockDim.x + threadIdx.x;
  if (i >= NNODES * C) return;
  int n = i / C, c = i - n * C;
  const float* a = accum + (size_t)n * H * C + c;
  float v = 0.f;
  for (int h = 0; h < H; ++h) v += a[h * C];
  v = v / (float)H + bias[c];
  if (doRelu) v = fmaxf(v, 0.f);
  out[i] = v;
}

__global__ void k_logsm(const float* __restrict__ in, float* __restrict__ out) {
  int n = blockIdx.x * blockDim.x + threadIdx.x;
  if (n >= NNODES) return;
  float z0 = in[2 * n], z1 = in[2 * n + 1];
  float mm = fmaxf(z0, z1);
  float l = logf(expf(z0 - mm) + expf(z1 - mm)) + mm;
  out[2 * n] = z0 - l;
  out[2 * n + 1] = z1 - l;
}

// ----------------------------------------------------------------- launcher
static inline int cdiv(long long a, int b) { return (int)((a + b - 1) / b); }

extern "C" void kernel_launch(void* const* d_in, const int* in_sizes, int n_in,
                              void* d_out, int out_size, void* d_ws, size_t ws_size,
                              hipStream_t stream) {
  const float* x   = (const float*)d_in[0];
  const int*   ei  = (const int*)d_in[1];          // edge_index, [2,E] flat
  const float* W1  = (const float*)d_in[2];
  const float* a1s = (const float*)d_in[3];
  const float* a1d = (const float*)d_in[4];
  const float* b1  = (const float*)d_in[5];
  const float* W2  = (const float*)d_in[6];
  const float* a2s = (const float*)d_in[7];
  const float* a2d = (const float*)d_in[8];
  const float* b2  = (const float*)d_in[9];
  const float* W3  = (const float*)d_in[10];
  const float* a3s = (const float*)d_in[11];
  const float* a3d = (const float*)d_in[12];
  const float* b3  = (const float*)d_in[13];
  const float* W4  = (const float*)d_in[14];
  const float* a4s = (const float*)d_in[15];
  const float* a4d = (const float*)d_in[16];
  const float* b4  = (const float*)d_in[17];

  char* p = (char*)d_ws;
  auto alloc = [&](size_t bytes) -> char* {
    char* r = p; p += (bytes + 255) & ~(size_t)255; return r;
  };
  int*      src   = (int*)alloc((size_t)ETOT * 4);
  int*      dst   = (int*)alloc((size_t)ETOT * 4);
  _Float16* xh    = (_Float16*)alloc((size_t)NNODES * NFEAT * 2);
  _Float16* wp    = (_Float16*)alloc(4 * 3 * 32 * 16 * 2);
  float*    feat  = (float*)alloc((size_t)NNODES * 96 * 4);   // post-GEMM [N,H*C]
  float*    accum = (float*)alloc((size_t)NNODES * 96 * 4);   // scatter target
  float*    es    = (float*)alloc((size_t)NNODES * 6 * 4);
  float*    ed    = (float*)alloc((size_t)NNODES * 6 * 4);
  float*    mbuf  = (float*)alloc((size_t)NNODES * 6 * 4);
  float*    sbuf  = (float*)alloc((size_t)NNODES * 6 * 4);
  float*    bufA  = (float*)alloc((size_t)NNODES * 16 * 4);
  float*    bufB  = (float*)alloc((size_t)NNODES * 16 * 4);

  k_edges<<<cdiv(ETOT, TPB), TPB, 0, stream>>>(ei, src, dst);
  k_f32tof16<<<cdiv((long long)NNODES * NFEAT, TPB), TPB, 0, stream>>>(x, xh, NNODES * NFEAT);
  k_packW1<<<cdiv(4 * 3 * 32 * 16, TPB), TPB, 0, stream>>>(W1, wp);
  k_gemm_wmma_l1<<<cdiv((long long)(NNODES / 16) * 32, TPB), TPB, 0, stream>>>(xh, wp, feat);

  auto attn = [&](const float* as_, const float* ad_, const float* bias,
                  int H, int C, float* outbuf, int doRelu) {
    k_scores<<<cdiv((long long)NNODES * H, TPB), TPB, 0, stream>>>(feat, as_, ad_, es, ed, mbuf, sbuf, H, C);
    k_zero<<<cdiv((long long)NNODES * H * C, TPB), TPB, 0, stream>>>(accum, NNODES * H * C);
    k_edge_max<<<cdiv(ETOT, TPB), TPB, 0, stream>>>(src, dst, es, ed, mbuf, H);
    k_edge_sum<<<cdiv(ETOT, TPB), TPB, 0, stream>>>(src, dst, es, ed, mbuf, sbuf, H);
    k_edge_scatter<<<cdiv((long long)ETOT * H, TPB), TPB, 0, stream>>>(src, dst, es, ed, mbuf, sbuf, feat, accum, H, C);
    k_head_mean<<<cdiv((long long)NNODES * C, TPB), TPB, 0, stream>>>(accum, bias, outbuf, H, C, doRelu);
  };

  // layer 1: 128 -> (6 heads x 8), relu
  attn(a1s, a1d, b1, 6, 8, bufA, 1);
  // layer 2: 8 -> (6 x 16), relu
  k_gemm_small<<<cdiv((long long)NNODES * 96, TPB), TPB, 0, stream>>>(bufA, W2, feat, 8, 96);
  attn(a2s, a2d, b2, 6, 16, bufB, 1);
  // layer 3: 16 -> (1 x 8), relu
  k_gemm_small<<<cdiv((long long)NNODES * 8, TPB), TPB, 0, stream>>>(bufB, W3, feat, 16, 8);
  attn(a3s, a3d, b3, 1, 8, bufA, 1);
  // layer 4: 8 -> (1 x 2), no relu
  k_gemm_small<<<cdiv((long long)NNODES * 2, TPB), TPB, 0, stream>>>(bufA, W4, feat, 8, 2);
  attn(a4s, a4d, b4, 1, 2, bufB, 0);

  k_logsm<<<cdiv(NNODES, TPB), TPB, 0, stream>>>(bufB, (float*)d_out);
}
